// MoEMLP_82617990905863
// MI455X (gfx1250) — matrix operand order repaired and checked by reading, not compiled, not run on hardware
//
#include <hip/hip_runtime.h>
#include <hip/hip_bf16.h>

// ---------------- problem constants ----------------
#define T_TOK   8192      // B*S
#define DIM     1024
#define NEXP    8
#define TOPK    2
#define MOED    2048
#define TK      (T_TOK * TOPK)        // 16384 dispatch rows
#define BM      128
#define KT      32
#define PADMAX  (TK + NEXP * BM)      // 17408 worst-case padded rows
#define ROWTILES (PADMAX / BM)        // 136

typedef __attribute__((ext_vector_type(16))) __bf16 v16bf;
typedef __attribute__((ext_vector_type(8)))  float  v8f;

union ABfrag { v16bf v; uint4 q[2]; };

__device__ __forceinline__ unsigned short f2bf(float f) {
    union { float f; unsigned u; } v; v.f = f;
    unsigned r = v.u + 0x7FFFu + ((v.u >> 16) & 1u);   // round-to-nearest-even
    return (unsigned short)(r >> 16);
}

// ---------------- 0) weight convert + transpose: [K][N] f32 -> [N][K] bf16 ----------------
__global__ __launch_bounds__(256) void k_convert_tr(
    const float* __restrict__ in, unsigned short* __restrict__ out, int K, int N)
{
    __shared__ unsigned short tile[32][36];   // padded; keeps 8B alignment for uint2
    const float* ine = in + (size_t)blockIdx.z * K * N;
    unsigned short* oute = out + (size_t)blockIdx.z * K * N;
    int n0 = blockIdx.x * 32, k0 = blockIdx.y * 32;
    int tid = threadIdx.x;
    {
        int r = tid >> 3, c4 = (tid & 7) * 4;            // r: k-row, c4: n-col group
        float4 v = *(const float4*)(ine + (size_t)(k0 + r) * N + n0 + c4);
        tile[c4 + 0][r] = f2bf(v.x);
        tile[c4 + 1][r] = f2bf(v.y);
        tile[c4 + 2][r] = f2bf(v.z);
        tile[c4 + 3][r] = f2bf(v.w);
    }
    __syncthreads();
    {
        int n = tid >> 3, s4 = (tid & 7) * 4;            // write 4 bf16 along k
        *(uint2*)(oute + (size_t)(n0 + n) * K + k0 + s4) = *(const uint2*)&tile[n][s4];
    }
}

// ---------------- 1) router + top2 + softmax ----------------
__global__ __launch_bounds__(256) void k_router(
    const float* __restrict__ x, const float* __restrict__ W,
    int* __restrict__ counts, int* __restrict__ tke, float* __restrict__ tkw)
{
    int wave = threadIdx.x >> 5, lane = threadIdx.x & 31;
    int t = blockIdx.x * 8 + wave;
    if (t >= T_TOK) return;
    float acc[NEXP];
    #pragma unroll
    for (int e = 0; e < NEXP; ++e) acc[e] = 0.f;
    const float* xr = x + (size_t)t * DIM;
    for (int d = lane; d < DIM; d += 32) {
        float xv = xr[d];
        const float* wr = W + (size_t)d * NEXP;
        #pragma unroll
        for (int e = 0; e < NEXP; ++e) acc[e] += xv * wr[e];
    }
    #pragma unroll
    for (int e = 0; e < NEXP; ++e) {
        #pragma unroll
        for (int off = 16; off >= 1; off >>= 1)
            acc[e] += __shfl_xor(acc[e], off, 32);
    }
    if (lane == 0) {
        int i0 = 0; float m0 = acc[0];
        #pragma unroll
        for (int e = 1; e < NEXP; ++e) if (acc[e] > m0) { m0 = acc[e]; i0 = e; }
        int i1 = -1; float m1 = -3.4e38f;
        #pragma unroll
        for (int e = 0; e < NEXP; ++e) if (e != i0 && acc[e] > m1) { m1 = acc[e]; i1 = e; }
        float e1 = __expf(m1 - m0);
        float s  = 1.f + e1;
        tke[2 * t]     = i0;  tkw[2 * t]     = 1.f / s;
        tke[2 * t + 1] = i1;  tkw[2 * t + 1] = e1 / s;
        atomicAdd(&counts[i0], 1);
        atomicAdd(&counts[i1], 1);
    }
}

// ---------------- 2) metadata ----------------
__global__ void k_init(int* counts, int* fill) {
    if (threadIdx.x < NEXP) { counts[threadIdx.x] = 0; fill[threadIdx.x] = 0; }
}
__global__ void k_offsets(const int* __restrict__ counts, int* __restrict__ offs) {
    if (threadIdx.x == 0) {
        int o = 0;
        for (int e = 0; e < NEXP; ++e) { offs[e] = o; o += ((counts[e] + BM - 1) / BM) * BM; }
        offs[NEXP] = o;
    }
}

// ---------------- 3) scatter (grouped layout + inverse map) ----------------
__global__ __launch_bounds__(256) void k_scatter(
    const int* __restrict__ tke, const float* __restrict__ tkw,
    const int* __restrict__ offs, int* __restrict__ fill,
    int* __restrict__ tok, float* __restrict__ wsrt, int* __restrict__ inv)
{
    int idx = blockIdx.x * 256 + threadIdx.x;
    if (idx >= TK) return;
    int e = tke[idx];
    int pos = offs[e] + atomicAdd(&fill[e], 1);
    tok[pos]  = idx >> 1;
    wsrt[pos] = tkw[idx];
    inv[idx]  = pos;
}

// ---------------- 4) gather tokens -> bf16 grouped buffer ----------------
__global__ __launch_bounds__(256) void k_gather(
    const float* __restrict__ x, const int* __restrict__ tok,
    const int* __restrict__ offs, const int* __restrict__ counts,
    unsigned short* __restrict__ xg)
{
    int pos = blockIdx.x;
    if (pos >= offs[NEXP]) return;
    int e = 0;
    #pragma unroll
    for (int i = 0; i < NEXP; ++i) if (pos >= offs[i + 1]) e = i + 1;
    unsigned short* dst = xg + (size_t)pos * DIM;
    int i4 = threadIdx.x * 4;                        // 256 thr * 4 = 1024
    if (pos - offs[e] < counts[e]) {
        const float* src = x + (size_t)tok[pos] * DIM;
        float4 v = *(const float4*)(src + i4);
        uint2 o;
        o.x = (unsigned)f2bf(v.x) | ((unsigned)f2bf(v.y) << 16);
        o.y = (unsigned)f2bf(v.z) | ((unsigned)f2bf(v.w) << 16);
        *(uint2*)(dst + i4) = o;
    } else {
        uint2 z; z.x = 0u; z.y = 0u;
        *(uint2*)(dst + i4) = z;
    }
}

// ---------------- 5) grouped GEMM1 + fused SwiGLU (ping-pong double buffer) ----------------
__global__ __launch_bounds__(256) void k_gemm1_swiglu(
    const unsigned short* __restrict__ xg, const unsigned short* __restrict__ w13t,
    unsigned short* __restrict__ hidden, const int* __restrict__ offs)
{
    __shared__ unsigned short sMem[4 * BM * KT];   // A0,A1,B0,B1
    int row0 = blockIdx.y * BM;
    if (row0 >= offs[NEXP]) return;
    int e = 0;
    #pragma unroll
    for (int i = 0; i < NEXP; ++i) if (row0 >= offs[i + 1]) e = i + 1;
    const unsigned short* wt = w13t + (size_t)e * (2 * MOED) * DIM;
    int n0 = blockIdx.x * 64;
    int tid = threadIdx.x;
    int wave = tid >> 5, lane = tid & 31;
    int wm = wave >> 1, wn = wave & 1;
    int g = lane >> 4, nl = lane & 15;

    int lr = tid >> 1, lh = (tid & 1) * 16;          // lr: row / tile-col, lh: k-half
    const unsigned short* aSrc = xg + (size_t)(row0 + lr) * DIM + lh;
    int q = lr >> 4, rr = lr & 15;
    int bn = (q & 1) ? (MOED + n0 + (q >> 1) * 16 + rr) : (n0 + (q >> 1) * 16 + rr);
    const unsigned short* bSrc = wt + (size_t)bn * DIM + lh;

    unsigned short* curA = sMem;
    unsigned short* nxtA = sMem + BM * KT;
    unsigned short* curB = sMem + 2 * BM * KT;
    unsigned short* nxtB = sMem + 3 * BM * KT;
    int ldOff = lr * KT + lh;

    v8f acc[2][4];
    #pragma unroll
    for (int s = 0; s < 2; ++s)
        #pragma unroll
        for (int j = 0; j < 4; ++j)
            acc[s][j] = (v8f){0.f,0.f,0.f,0.f,0.f,0.f,0.f,0.f};

    auto compute_tile = [&](const unsigned short* cA, const unsigned short* cB) {
        ABfrag a[2];
        #pragma unroll
        for (int s = 0; s < 2; ++s) {
            int ra = wm * 32 + s * 16 + nl;
            a[s].q[0] = *(const uint4*)(cA + ra * KT + g * 8);       // j=0..3
            a[s].q[1] = *(const uint4*)(cA + ra * KT + 16 + g * 8);  // j=4..7
        }
        #pragma unroll
        for (int j4 = 0; j4 < 4; ++j4) {
            int tc = (wn * 4 + j4) * 16 + nl;
            ABfrag b;
            b.q[0] = *(const uint4*)(cB + tc * KT + g * 8);
            b.q[1] = *(const uint4*)(cB + tc * KT + 16 + g * 8);
            acc[0][j4] = __builtin_amdgcn_wmma_f32_16x16x32_bf16(
                false, a[0].v, false, b.v, (short)0, acc[0][j4], false, false);
            acc[1][j4] = __builtin_amdgcn_wmma_f32_16x16x32_bf16(
                false, a[1].v, false, b.v, (short)0, acc[1][j4], false, false);
        }
    };

    // prologue: stage tile 0
    {
        uint4 a0 = *(const uint4*)aSrc, a1 = *(const uint4*)(aSrc + 8);
        uint4 b0 = *(const uint4*)bSrc, b1 = *(const uint4*)(bSrc + 8);
        ((uint4*)(curA + ldOff))[0] = a0; ((uint4*)(curA + ldOff))[1] = a1;
        ((uint4*)(curB + ldOff))[0] = b0; ((uint4*)(curB + ldOff))[1] = b1;
    }

    const int NIT = DIM / KT;   // 32
    #pragma unroll 1
    for (int it = 0; it < NIT - 1; ++it) {
        __syncthreads();
        int ko = (it + 1) * KT;
        uint4 a0 = *(const uint4*)(aSrc + ko), a1 = *(const uint4*)(aSrc + ko + 8);
        uint4 b0 = *(const uint4*)(bSrc + ko), b1 = *(const uint4*)(bSrc + ko + 8);
        compute_tile(curA, curB);
        ((uint4*)(nxtA + ldOff))[0] = a0; ((uint4*)(nxtA + ldOff))[1] = a1;
        ((uint4*)(nxtB + ldOff))[0] = b0; ((uint4*)(nxtB + ldOff))[1] = b1;
        unsigned short* t;
        t = curA; curA = nxtA; nxtA = t;
        t = curB; curB = nxtB; nxtB = t;
    }
    __syncthreads();
    compute_tile(curA, curB);

    // SwiGLU in-register: pairs (even=gate, odd=up) share row/col layout
    #pragma unroll
    for (int s = 0; s < 2; ++s) {
        #pragma unroll
        for (int p = 0; p < 2; ++p) {
            v8f ga = acc[s][2 * p], ua = acc[s][2 * p + 1];
            int col = n0 + (wn * 2 + p) * 16 + nl;
            #pragma unroll
            for (int v = 0; v < 8; ++v) {
                int row = row0 + wm * 32 + s * 16 + v + 8 * g;
                float gt = ga[v];
                float h = gt * __builtin_amdgcn_rcpf(1.f + __expf(-gt)) * ua[v];
                hidden[(size_t)row * MOED + col] = f2bf(h);
            }
        }
    }
}

// ---------------- 6) grouped GEMM2 (ping-pong double buffer) ----------------
__global__ __launch_bounds__(256) void k_gemm2(
    const unsigned short* __restrict__ hidden, const unsigned short* __restrict__ w2t,
    float* __restrict__ od, const int* __restrict__ offs)
{
    __shared__ unsigned short sMem[4 * BM * KT];
    int row0 = blockIdx.y * BM;
    if (row0 >= offs[NEXP]) return;
    int e = 0;
    #pragma unroll
    for (int i = 0; i < NEXP; ++i) if (row0 >= offs[i + 1]) e = i + 1;
    const unsigned short* wt = w2t + (size_t)e * DIM * MOED;
    int n0 = blockIdx.x * BM;
    int tid = threadIdx.x;
    int wave = tid >> 5, lane = tid & 31;
    int wm = wave >> 1, wn = wave & 1;
    int g = lane >> 4, nl = lane & 15;

    int lr = tid >> 1, lh = (tid & 1) * 16;
    const unsigned short* aSrc = hidden + (size_t)(row0 + lr) * MOED + lh;
    const unsigned short* bSrc = wt + (size_t)(n0 + lr) * MOED + lh;

    unsigned short* curA = sMem;
    unsigned short* nxtA = sMem + BM * KT;
    unsigned short* curB = sMem + 2 * BM * KT;
    unsigned short* nxtB = sMem + 3 * BM * KT;
    int ldOff = lr * KT + lh;

    v8f acc[2][4];
    #pragma unroll
    for (int s = 0; s < 2; ++s)
        #pragma unroll
        for (int j = 0; j < 4; ++j)
            acc[s][j] = (v8f){0.f,0.f,0.f,0.f,0.f,0.f,0.f,0.f};

    auto compute_tile = [&](const unsigned short* cA, const unsigned short* cB) {
        ABfrag a[2];
        #pragma unroll
        for (int s = 0; s < 2; ++s) {
            int ra = wm * 32 + s * 16 + nl;
            a[s].q[0] = *(const uint4*)(cA + ra * KT + g * 8);
            a[s].q[1] = *(const uint4*)(cA + ra * KT + 16 + g * 8);
        }
        #pragma unroll
        for (int j4 = 0; j4 < 4; ++j4) {
            int tc = (wn * 4 + j4) * 16 + nl;
            ABfrag b;
            b.q[0] = *(const uint4*)(cB + tc * KT + g * 8);
            b.q[1] = *(const uint4*)(cB + tc * KT + 16 + g * 8);
            acc[0][j4] = __builtin_amdgcn_wmma_f32_16x16x32_bf16(
                false, a[0].v, false, b.v, (short)0, acc[0][j4], false, false);
            acc[1][j4] = __builtin_amdgcn_wmma_f32_16x16x32_bf16(
                false, a[1].v, false, b.v, (short)0, acc[1][j4], false, false);
        }
    };

    {
        uint4 a0 = *(const uint4*)aSrc, a1 = *(const uint4*)(aSrc + 8);
        uint4 b0 = *(const uint4*)bSrc, b1 = *(const uint4*)(bSrc + 8);
        ((uint4*)(curA + ldOff))[0] = a0; ((uint4*)(curA + ldOff))[1] = a1;
        ((uint4*)(curB + ldOff))[0] = b0; ((uint4*)(curB + ldOff))[1] = b1;
    }

    const int NIT = MOED / KT;  // 64
    #pragma unroll 1
    for (int it = 0; it < NIT - 1; ++it) {
        __syncthreads();
        int ko = (it + 1) * KT;
        uint4 a0 = *(const uint4*)(aSrc + ko), a1 = *(const uint4*)(aSrc + ko + 8);
        uint4 b0 = *(const uint4*)(bSrc + ko), b1 = *(const uint4*)(bSrc + ko + 8);
        compute_tile(curA, curB);
        ((uint4*)(nxtA + ldOff))[0] = a0; ((uint4*)(nxtA + ldOff))[1] = a1;
        ((uint4*)(nxtB + ldOff))[0] = b0; ((uint4*)(nxtB + ldOff))[1] = b1;
        unsigned short* t;
        t = curA; curA = nxtA; nxtA = t;
        t = curB; curB = nxtB; nxtB = t;
    }
    __syncthreads();
    compute_tile(curA, curB);

    #pragma unroll
    for (int s = 0; s < 2; ++s) {
        #pragma unroll
        for (int j4 = 0; j4 < 4; ++j4) {
            int col = n0 + (wn * 4 + j4) * 16 + nl;
            #pragma unroll
            for (int v = 0; v < 8; ++v) {
                int row = row0 + wm * 32 + s * 16 + v + 8 * g;
                od[(size_t)row * DIM + col] = acc[s][j4][v];
            }
        }
    }
}

// ---------------- 7) deterministic combine ----------------
__global__ __launch_bounds__(256) void k_combine(
    const float* __restrict__ od, const float* __restrict__ wsrt,
    const int* __restrict__ inv, float* __restrict__ out)
{
    int t = blockIdx.x;
    int c = threadIdx.x * 4;                         // 256 thr * 4 = 1024
    int p0 = inv[2 * t], p1 = inv[2 * t + 1];
    float w0 = wsrt[p0], w1 = wsrt[p1];
    float4 a = *(const float4*)(od + (size_t)p0 * DIM + c);
    float4 b = *(const float4*)(od + (size_t)p1 * DIM + c);
    float4 o;
    o.x = w0 * a.x + w1 * b.x;
    o.y = w0 * a.y + w1 * b.y;
    o.z = w0 * a.z + w1 * b.z;
    o.w = w0 * a.w + w1 * b.w;
    *(float4*)(out + (size_t)t * DIM + c) = o;
}

// ---------------- host launcher ----------------
extern "C" void kernel_launch(void* const* d_in, const int* in_sizes, int n_in,
                              void* d_out, int out_size, void* d_ws, size_t ws_size,
                              hipStream_t stream) {
    (void)in_sizes; (void)n_in; (void)out_size; (void)ws_size;
    const float* x   = (const float*)d_in[0];
    const float* Wr  = (const float*)d_in[1];
    const float* w13 = (const float*)d_in[2];
    const float* w2  = (const float*)d_in[3];
    float* out = (float*)d_out;

    char* p = (char*)d_ws;
    auto carve = [&](size_t bytes) -> void* {
        void* r = (void*)p;
        p += (bytes + 255) & ~(size_t)255;
        return r;
    };
    int*   counts = (int*)carve(NEXP * 4);
    int*   fill   = (int*)carve(NEXP * 4);
    int*   offs   = (int*)carve((NEXP + 1) * 4);
    int*   tke    = (int*)carve((size_t)TK * 4);
    float* tkw    = (float*)carve((size_t)TK * 4);
    int*   tok    = (int*)carve((size_t)PADMAX * 4);
    float* wsrt   = (float*)carve((size_t)PADMAX * 4);
    int*   inv    = (int*)carve((size_t)TK * 4);
    unsigned short* xg   = (unsigned short*)carve((size_t)PADMAX * DIM * 2);
    unsigned short* hid  = (unsigned short*)carve((size_t)PADMAX * MOED * 2);
    float*          od   = (float*)carve((size_t)PADMAX * DIM * 4);
    unsigned short* w13t = (unsigned short*)carve((size_t)NEXP * (2 * MOED) * DIM * 2);
    unsigned short* w2t  = (unsigned short*)carve((size_t)NEXP * DIM * MOED * 2);

    // one-shot weight convert+transpose (bf16, n-major)
    dim3 gcw13(2 * MOED / 32, DIM / 32, NEXP);       // in [1024][4096] -> out [4096][1024]
    k_convert_tr<<<gcw13, 256, 0, stream>>>(w13, w13t, DIM, 2 * MOED);
    dim3 gcw2(DIM / 32, MOED / 32, NEXP);            // in [2048][1024] -> out [1024][2048]
    k_convert_tr<<<gcw2, 256, 0, stream>>>(w2, w2t, MOED, DIM);

    k_init<<<1, 32, 0, stream>>>(counts, fill);
    k_router<<<T_TOK / 8, 256, 0, stream>>>(x, Wr, counts, tke, tkw);
    k_offsets<<<1, 1, 0, stream>>>(counts, offs);
    k_scatter<<<TK / 256, 256, 0, stream>>>(tke, tkw, offs, fill, tok, wsrt, inv);
    k_gather<<<PADMAX, 256, 0, stream>>>(x, tok, offs, counts, xg);
    dim3 g1(2 * MOED / 128, ROWTILES);   // 32 x 136
    k_gemm1_swiglu<<<g1, 256, 0, stream>>>(xg, w13t, hid, offs);
    dim3 g2(DIM / BM, ROWTILES);         // 8 x 136
    k_gemm2<<<g2, 256, 0, stream>>>(hid, w2t, od, offs);
    k_combine<<<T_TOK, 256, 0, stream>>>(od, wsrt, inv, out);
}
